// StyleCrossAttention_86784109183126
// MI455X (gfx1250) — matrix operand order repaired and checked by reading, not compiled, not run on hardware
//
#include <hip/hip_runtime.h>
#include <math.h>

// ---------------------------------------------------------------------------
// Types
// ---------------------------------------------------------------------------
typedef __attribute__((ext_vector_type(16))) __bf16 bf16x16;
typedef __attribute__((ext_vector_type(8)))  __bf16 bf16x8;
typedef __attribute__((ext_vector_type(8)))  float  f32x8;

#define DEV __device__ __forceinline__

// Build a 16-element bf16 fragment from two contiguous 16-byte chunks.
DEV bf16x16 ld_frag(const __bf16* p0, const __bf16* p1) {
  bf16x16 r;
  *reinterpret_cast<bf16x8*>(&r)       = *reinterpret_cast<const bf16x8*>(p0);
  *(reinterpret_cast<bf16x8*>(&r) + 1) = *reinterpret_cast<const bf16x8*>(p1);
  return r;
}

DEV f32x8 wmma_bf16(bf16x16 a, bf16x16 b, f32x8 c) {
  return __builtin_amdgcn_wmma_f32_16x16x32_bf16(false, a, false, b, (short)0, c,
                                                 false, false);
}

template <int IMM>
DEV float swz(float x) {
  return __int_as_float(__builtin_amdgcn_ds_swizzle(__float_as_int(x), IMM));
}

DEV float wave_sum32(float s) {
  s += swz<0x041f>(s);  // xor 1
  s += swz<0x081f>(s);  // xor 2
  s += swz<0x101f>(s);  // xor 4
  s += swz<0x201f>(s);  // xor 8
  s += swz<0x401f>(s);  // xor 16
  return s;
}

// ---------------------------------------------------------------------------
// fp32 -> bf16 convert (grid-stride)
// ---------------------------------------------------------------------------
__global__ __launch_bounds__(256) void cvt_bf16(const float* __restrict__ x,
                                                __bf16* __restrict__ y, long n) {
  long i = (long)blockIdx.x * blockDim.x + threadIdx.x;
  const long stride = (long)gridDim.x * blockDim.x;
  for (; i < n; i += stride) y[i] = (__bf16)x[i];
}

// ---------------------------------------------------------------------------
// bool mask -> additive bias in log2-softmax domain (0 or -1e30)
// ---------------------------------------------------------------------------
__global__ __launch_bounds__(256) void mask_bias(
    const unsigned char* __restrict__ m, float* __restrict__ bias, int n) {
  const int i = blockIdx.x * blockDim.x + threadIdx.x;
  if (i < n) bias[i] = m[i] ? 0.0f : -1.0e30f;
}

// ---------------------------------------------------------------------------
// Generic bf16 WMMA GEMM:  C[M,N] = A[M,K] * W[K,N] + bias, with fused epilogue
// Block = 256 threads = 8 waves, each wave does a 32x64 tile -> 128x128/block.
// ---------------------------------------------------------------------------
enum { EPI_F32_BF16 = 0, EPI_QT = 1, EPI_BF16 = 2, EPI_RESID = 3, EPI_GELU = 4 };

template <int EPI>
__global__ __launch_bounds__(256) void gemm_wmma(
    const __bf16* __restrict__ A, const __bf16* __restrict__ W,
    const float* __restrict__ bias, const float* __restrict__ resid,
    float* __restrict__ outF, __bf16* __restrict__ outB, int M, int N, int K) {
  const int lane = threadIdx.x & 31;
  const int wave = threadIdx.x >> 5;
  const int m0 = blockIdx.x * 128 + (wave >> 1) * 32;
  const int n0 = blockIdx.y * 128 + (wave & 1) * 64;
  const int half = lane >> 4, l15 = lane & 15;

  f32x8 acc[2][4];
#pragma unroll
  for (int mt = 0; mt < 2; ++mt)
#pragma unroll
    for (int nt = 0; nt < 4; ++nt)
#pragma unroll
      for (int i = 0; i < 8; ++i) acc[mt][nt][i] = 0.f;

  for (int k0 = 0; k0 < K; k0 += 32) {
    bf16x16 af[2], bfr[4];
#pragma unroll
    for (int mt = 0; mt < 2; ++mt) {
      const __bf16* p = A + (size_t)(m0 + mt * 16 + l15) * K + (k0 + half * 8);
      af[mt] = ld_frag(p, p + 16);
    }
#pragma unroll
    for (int nt = 0; nt < 4; ++nt) {
      const __bf16* p = W + (size_t)k0 * N + (n0 + nt * 16 + l15);
      for (int s = 0; s < 16; ++s)
        bfr[nt][s] = p[(size_t)((s & 7) + half * 8 + (s >> 3) * 16) * N];
    }
#pragma unroll
    for (int mt = 0; mt < 2; ++mt)
#pragma unroll
      for (int nt = 0; nt < 4; ++nt)
        acc[mt][nt] = wmma_bf16(af[mt], bfr[nt], acc[mt][nt]);
  }

#pragma unroll
  for (int mt = 0; mt < 2; ++mt) {
    const int mbase = m0 + mt * 16 + half * 8;
#pragma unroll
    for (int nt = 0; nt < 4; ++nt) {
      const int n = n0 + nt * 16 + l15;
      const float bv = bias[n];
      if (EPI == EPI_QT) {
        // transposed bf16 store: qT[b][n][m%2048], 8 consecutive m -> 16B chunk
        bf16x8 pk;
#pragma unroll
        for (int i = 0; i < 8; ++i) pk[i] = (__bf16)(acc[mt][nt][i] + bv);
        const size_t off = (size_t)(mbase >> 11) * (512u * 2048u) +
                           (size_t)n * 2048u + (size_t)(mbase & 2047);
        *reinterpret_cast<bf16x8*>(outB + off) = pk;
      } else {
#pragma unroll
        for (int i = 0; i < 8; ++i) {
          const size_t off = (size_t)(mbase + i) * N + n;
          const float v = acc[mt][nt][i] + bv;
          if (EPI == EPI_RESID) {
            outF[off] = v + resid[off];
          } else if (EPI == EPI_GELU) {
            outB[off] = (__bf16)(0.5f * v * (1.0f + erff(v * 0.70710678118f)));
          } else if (EPI == EPI_BF16) {
            outB[off] = (__bf16)v;
          } else {  // EPI_F32_BF16
            outF[off] = v;
            outB[off] = (__bf16)v;
          }
        }
      }
    }
  }
}

// ---------------------------------------------------------------------------
// Attention: S^T = K * Q^T  (queries along lanes -> per-query stats in-lane)
// log2-domain softmax: t = s*scale*log2e + maskbias; p = exp2(t - shift).
// 3 passes over keys: max, sum-exp, emit (attn fp32 to d_out, O = P*V).
// grid = (NS/128, HEADS, B), block = 256 (8 waves x 16 queries each).
// ---------------------------------------------------------------------------
DEV f32x8 score_tile(const __bf16* kBb, const bf16x16* bq, int kt, int l15,
                     int half) {
  f32x8 s;
#pragma unroll
  for (int i = 0; i < 8; ++i) s[i] = 0.f;
#pragma unroll
  for (int c = 0; c < 2; ++c) {
    const __bf16* kp = kBb + (size_t)(kt + l15) * 512 + c * 32 + half * 8;
    s = wmma_bf16(ld_frag(kp, kp + 16), bq[c], s);
  }
  return s;
}

__global__ __launch_bounds__(256) void attn_kernel(
    const __bf16* __restrict__ qT, const __bf16* __restrict__ kB,
    const __bf16* __restrict__ vB, const float* __restrict__ mbias,
    float* __restrict__ attnOut, __bf16* __restrict__ fused) {
  __shared__ __attribute__((aligned(16))) __bf16 pstage[8][16][32];

  const int lane = threadIdx.x & 31, wave = threadIdx.x >> 5;
  const int half = lane >> 4, l15 = lane & 15;
  const int b = blockIdx.z, h = blockIdx.y;
  const int q0 = blockIdx.x * 128 + wave * 16;
  // scale * log2(e): softmax done in exp2 domain (hardware v_exp_f32)
  const float scl2 = 0.125f * 1.4426950408889634f;

  const __bf16* qTb = qT + ((size_t)b * 512 + h * 64) * 2048 + q0;
  const __bf16* kBb = kB + (size_t)b * 1024 * 512 + h * 64;
  const __bf16* vBb = vB + (size_t)b * 1024 * 512 + h * 64;
  const float* mbb = mbias + b * 1024;

  // Q^T B-fragments (reused across all key tiles): lane = query, elems = d
  bf16x16 bq[2];
#pragma unroll
  for (int c = 0; c < 2; ++c) {
    const __bf16* p = qTb + (size_t)(c * 32) * 2048 + l15;
    for (int s = 0; s < 16; ++s)
      bq[c][s] = p[(size_t)((s & 7) + half * 8 + (s >> 3) * 16) * 2048];
  }

  // ---- pass 1: per-query max of t = s*scl2 + bias ----
  float qmax = -3.4e38f;
  for (int kt = 0; kt < 1024; kt += 16) {
    const f32x8 s = score_tile(kBb, bq, kt, l15, half);
    const float4* bp = reinterpret_cast<const float4*>(mbb + kt + half * 8);
    const float4 b0 = bp[0], b1 = bp[1];
    const float bb[8] = {b0.x, b0.y, b0.z, b0.w, b1.x, b1.y, b1.z, b1.w};
#pragma unroll
    for (int i = 0; i < 8; ++i) qmax = fmaxf(qmax, fmaf(s[i], scl2, bb[i]));
  }
  qmax = fmaxf(qmax, swz<0x401f>(qmax));  // combine the two key-halves

  // ---- pass 2: sum of exp2 ----
  float sum = 0.f;
  for (int kt = 0; kt < 1024; kt += 16) {
    const f32x8 s = score_tile(kBb, bq, kt, l15, half);
    const float4* bp = reinterpret_cast<const float4*>(mbb + kt + half * 8);
    const float4 b0 = bp[0], b1 = bp[1];
    const float bb[8] = {b0.x, b0.y, b0.z, b0.w, b1.x, b1.y, b1.z, b1.w};
#pragma unroll
    for (int i = 0; i < 8; ++i)
      sum += __builtin_amdgcn_exp2f(fmaf(s[i], scl2, bb[i]) - qmax);
  }
  sum += swz<0x401f>(sum);
  // p = exp2(t - qmax)/sum = exp2(t - shift), shift = qmax + log2(sum)
  const float shift = qmax + __builtin_amdgcn_logf(fmaxf(sum, 1e-30f));

  // ---- pass 3: emit attn + accumulate O = P * V ----
  f32x8 oacc[4];
#pragma unroll
  for (int d = 0; d < 4; ++d)
#pragma unroll
    for (int i = 0; i < 8; ++i) oacc[d][i] = 0.f;

  float* arow = attnOut + (((size_t)b * 8 + h) * 2048 + q0 + l15) * 1024;

  for (int kt = 0; kt < 1024; kt += 32) {
#pragma unroll
    for (int sub = 0; sub < 2; ++sub) {
      const int kb0 = kt + sub * 16;
      const f32x8 s = score_tile(kBb, bq, kb0, l15, half);
      const float4* bp = reinterpret_cast<const float4*>(mbb + kb0 + half * 8);
      const float4 b0 = bp[0], b1 = bp[1];
      const float bb[8] = {b0.x, b0.y, b0.z, b0.w, b1.x, b1.y, b1.z, b1.w};
      float p[8];
#pragma unroll
      for (int i = 0; i < 8; ++i)
        p[i] = __builtin_amdgcn_exp2f(fmaf(s[i], scl2, bb[i]) - shift);
      // coalesced fp32 probs to d_out (8 consecutive keys per lane)
      float4* ap = reinterpret_cast<float4*>(arow + kb0 + half * 8);
      ap[0] = make_float4(p[0], p[1], p[2], p[3]);
      ap[1] = make_float4(p[4], p[5], p[6], p[7]);
      // stage bf16 P tile in per-wave LDS (A-fragment source layout)
      bf16x8 pk;
#pragma unroll
      for (int i = 0; i < 8; ++i) pk[i] = (__bf16)p[i];
      *reinterpret_cast<bf16x8*>(&pstage[wave][l15][sub * 16 + half * 8]) = pk;
    }
    // P A-fragment (16 queries x 32 keys) from LDS, V B-fragments from global
    const __bf16* pp = &pstage[wave][l15][half * 8];
    const bf16x16 pa = ld_frag(pp, pp + 16);
#pragma unroll
    for (int d = 0; d < 4; ++d) {
      const __bf16* vp = vBb + (size_t)kt * 512 + d * 16 + l15;
      bf16x16 vf;
      for (int s = 0; s < 16; ++s)
        vf[s] = vp[(size_t)((s & 7) + half * 8 + (s >> 3) * 16) * 512];
      oacc[d] = wmma_bf16(pa, vf, oacc[d]);
    }
  }

#pragma unroll
  for (int d = 0; d < 4; ++d)
#pragma unroll
    for (int i = 0; i < 8; ++i)
      fused[(size_t)(b * 2048 + q0 + half * 8 + i) * 512 + h * 64 + d * 16 +
            l15] = (__bf16)oacc[d][i];
}

// ---------------------------------------------------------------------------
// LayerNorm over rows of 512 (one wave per row, exact two-pass mean/var)
// ---------------------------------------------------------------------------
__global__ __launch_bounds__(256) void ln_rows(const float* __restrict__ x,
                                               const float* __restrict__ g,
                                               const float* __restrict__ bt,
                                               float* __restrict__ outF,
                                               __bf16* __restrict__ outB) {
  const int lane = threadIdx.x & 31, wave = threadIdx.x >> 5;
  const size_t row = (size_t)blockIdx.x * 8 + wave;
  const float4* xr = reinterpret_cast<const float4*>(x + row * 512);
  float4 v[4];
  float s = 0.f;
#pragma unroll
  for (int j = 0; j < 4; ++j) {
    v[j] = xr[lane + 32 * j];
    s += v[j].x + v[j].y + v[j].z + v[j].w;
  }
  const float mu = wave_sum32(s) * (1.0f / 512.0f);
  float s2 = 0.f;
#pragma unroll
  for (int j = 0; j < 4; ++j) {
    const float dx = v[j].x - mu, dy = v[j].y - mu, dz = v[j].z - mu,
                dw = v[j].w - mu;
    s2 += dx * dx + dy * dy + dz * dz + dw * dw;
  }
  const float rs = rsqrtf(wave_sum32(s2) * (1.0f / 512.0f) + 1e-5f);
  float4* of = reinterpret_cast<float4*>(outF + row * 512);
#pragma unroll
  for (int j = 0; j < 4; ++j) {
    const int idx = (lane + 32 * j) * 4;
    float4 y;
    y.x = (v[j].x - mu) * rs * g[idx + 0] + bt[idx + 0];
    y.y = (v[j].y - mu) * rs * g[idx + 1] + bt[idx + 1];
    y.z = (v[j].z - mu) * rs * g[idx + 2] + bt[idx + 2];
    y.w = (v[j].w - mu) * rs * g[idx + 3] + bt[idx + 3];
    of[lane + 32 * j] = y;
    if (outB) {
      __bf16* ob = outB + row * 512 + idx;
      ob[0] = (__bf16)y.x;
      ob[1] = (__bf16)y.y;
      ob[2] = (__bf16)y.z;
      ob[3] = (__bf16)y.w;
    }
  }
}

// ---------------------------------------------------------------------------
// Host launcher
// ---------------------------------------------------------------------------
extern "C" void kernel_launch(void* const* d_in, const int* in_sizes, int n_in,
                              void* d_out, int out_size, void* d_ws,
                              size_t ws_size, hipStream_t stream) {
  (void)in_sizes; (void)n_in; (void)out_size; (void)ws_size;

  const float* src = (const float*)d_in[0];
  const float* sty = (const float*)d_in[1];
  const unsigned char* msk = (const unsigned char*)d_in[2];
  const float* Ws = (const float*)d_in[3];
  const float* bs = (const float*)d_in[4];
  const float* Wq = (const float*)d_in[5];
  const float* bq = (const float*)d_in[6];
  const float* Wk = (const float*)d_in[7];
  const float* bk = (const float*)d_in[8];
  const float* Wv = (const float*)d_in[9];
  const float* bv = (const float*)d_in[10];
  const float* Wo = (const float*)d_in[11];
  const float* bo = (const float*)d_in[12];
  const float* ln1w = (const float*)d_in[13];
  const float* ln1b = (const float*)d_in[14];
  const float* ln2w = (const float*)d_in[15];
  const float* ln2b = (const float*)d_in[16];
  const float* W1 = (const float*)d_in[17];
  const float* b1 = (const float*)d_in[18];
  const float* W2 = (const float*)d_in[19];
  const float* b2 = (const float*)d_in[20];

  constexpr long nB = 8, nS = 2048, nK = 1024, H = 512, MLPH = 1024, SD = 768;
  const long MQ = nB * nS;  // 16384
  const long MK = nB * nK;  // 8192

  char* ws = (char*)d_ws;
  size_t cur = 0;
  auto alloc = [&](size_t bytes) -> void* {
    void* p = ws + cur;
    cur = (cur + bytes + 255) & ~(size_t)255;
    return p;
  };

  __bf16* srcb  = (__bf16*)alloc(MQ * SD * 2);
  __bf16* styb  = (__bf16*)alloc(MK * SD * 2);
  __bf16* Wsb   = (__bf16*)alloc(SD * H * 2);
  __bf16* Wqb   = (__bf16*)alloc(H * H * 2);
  __bf16* Wkb   = (__bf16*)alloc(SD * H * 2);
  __bf16* Wvb   = (__bf16*)alloc(SD * H * 2);
  __bf16* Wob   = (__bf16*)alloc(H * H * 2);
  __bf16* W1b   = (__bf16*)alloc(H * MLPH * 2);
  __bf16* W2b   = (__bf16*)alloc(MLPH * H * 2);
  float*  mbias = (float*)alloc(MK * 4);
  float*  srchF = (float*)alloc(MQ * H * 4);
  __bf16* srchB = (__bf16*)alloc(MQ * H * 2);
  __bf16* qT    = (__bf16*)alloc(MQ * H * 2);   // [B][512][2048]
  __bf16* kbuf  = (__bf16*)alloc(MK * H * 2);
  __bf16* vbuf  = (__bf16*)alloc(MK * H * 2);
  __bf16* fusedb= (__bf16*)alloc(MQ * H * 2);
  float*  residf= (float*)alloc(MQ * H * 4);
  float*  hidF  = (float*)alloc(MQ * H * 4);
  __bf16* hidB  = (__bf16*)alloc(MQ * H * 2);
  __bf16* mlp1b = (__bf16*)alloc(MQ * MLPH * 2);

  float* outHidden = (float*)d_out;
  float* outAttn = outHidden + (size_t)MQ * H;  // 8,388,608 floats in

  // ---- fp32 -> bf16 conversions + mask bias ----
  cvt_bf16<<<2048, 256, 0, stream>>>(src, srcb, MQ * SD);
  cvt_bf16<<<2048, 256, 0, stream>>>(sty, styb, MK * SD);
  cvt_bf16<<<512, 256, 0, stream>>>(Ws, Wsb, SD * H);
  cvt_bf16<<<512, 256, 0, stream>>>(Wq, Wqb, H * H);
  cvt_bf16<<<512, 256, 0, stream>>>(Wk, Wkb, SD * H);
  cvt_bf16<<<512, 256, 0, stream>>>(Wv, Wvb, SD * H);
  cvt_bf16<<<512, 256, 0, stream>>>(Wo, Wob, H * H);
  cvt_bf16<<<512, 256, 0, stream>>>(W1, W1b, H * MLPH);
  cvt_bf16<<<512, 256, 0, stream>>>(W2, W2b, MLPH * H);
  mask_bias<<<32, 256, 0, stream>>>(msk, mbias, (int)MK);

  // ---- src_h = src @ Ws + bs  (fp32 + bf16) ----
  gemm_wmma<EPI_F32_BF16><<<dim3(128, 4), 256, 0, stream>>>(
      srcb, Wsb, bs, nullptr, srchF, srchB, 16384, 512, 768);
  // ---- qT = (src_h @ Wq + bq)^T  (bf16, [B][512][2048]) ----
  gemm_wmma<EPI_QT><<<dim3(128, 4), 256, 0, stream>>>(
      srchB, Wqb, bq, nullptr, nullptr, qT, 16384, 512, 512);
  // ---- k = style @ Wk + bk ; v = style @ Wv + bv  (bf16) ----
  gemm_wmma<EPI_BF16><<<dim3(64, 4), 256, 0, stream>>>(
      styb, Wkb, bk, nullptr, nullptr, kbuf, 8192, 512, 768);
  gemm_wmma<EPI_BF16><<<dim3(64, 4), 256, 0, stream>>>(
      styb, Wvb, bv, nullptr, nullptr, vbuf, 8192, 512, 768);

  // ---- attention: attn probs -> d_out, fused = attn @ v -> bf16 ----
  attn_kernel<<<dim3(16, 8, 8), 256, 0, stream>>>(qT, kbuf, vbuf, mbias,
                                                  outAttn, fusedb);

  // ---- resid1 = src_h + fused @ Wo + bo ; hidden = LN1(resid1) ----
  gemm_wmma<EPI_RESID><<<dim3(128, 4), 256, 0, stream>>>(
      fusedb, Wob, bo, srchF, residf, nullptr, 16384, 512, 512);
  ln_rows<<<2048, 256, 0, stream>>>(residf, ln1w, ln1b, hidF, hidB);

  // ---- mlp = gelu(hidden @ W1 + b1) @ W2 + b2 ; out = LN2(hidden + mlp) ----
  gemm_wmma<EPI_GELU><<<dim3(128, 8), 256, 0, stream>>>(
      hidB, W1b, b1, nullptr, nullptr, mlp1b, 16384, 1024, 512);
  gemm_wmma<EPI_RESID><<<dim3(128, 4), 256, 0, stream>>>(
      mlp1b, W2b, b2, hidF, residf, nullptr, 16384, 512, 1024);
  ln_rows<<<2048, 256, 0, stream>>>(residf, ln2w, ln2b, outHidden, nullptr);
}
